// STGAT_1872605741374
// MI455X (gfx1250) — matrix operand order repaired and checked by reading, not compiled, not run on hardware
//
#include <hip/hip_runtime.h>
#include <hip/hip_bf16.h>

typedef __attribute__((ext_vector_type(16))) __bf16 v16bf;
typedef __attribute__((ext_vector_type(8)))  __bf16 v8bf;
typedef __attribute__((ext_vector_type(8)))  float  v8f;

#define OBS 8
#define FUT 12
#define HT 32
#define HG 32
#define GG 1024
#define PP 64
#define NN (GG*PP)

__device__ __forceinline__ __bf16 f2bf(float x) { return (__bf16)x; }
__device__ __forceinline__ float sigf(float x) { return 1.0f / (1.0f + __expf(-x)); }

__device__ __forceinline__ v8f wmma_bf(v16bf a, v16bf b, v8f c) {
  return __builtin_amdgcn_wmma_f32_16x16x32_bf16(false, a, false, b, (short)0, c, false, false);
}

// Packed-B layout (contiguous k=16h+s, same order as load_a_frag): element (k,n) -> tile (k>>5)*NT+(n>>4),
// lane ((k>>4)&1)*16 + (n&15), vector element (k&15). One fragment = 32
// contiguous bytes per lane -> 2x ds_load_b128.
__device__ __forceinline__ int b_pack_off(int k, int n, int NT) {
  return ((((k >> 5) * NT + (n >> 4)) * 32) + (((k >> 4) & 1) * 16) + (n & 15)) * 16 + (k & 15);
}
__device__ __forceinline__ v16bf load_b_packed(const __bf16* Bp, int tile, int lane) {
  return *(const v16bf*)(Bp + ((size_t)tile * 32 + lane) * 16);
}
// A row-major [M][lda]; per-lane fragment = two contiguous 16B chunks
__device__ __forceinline__ v16bf load_a_frag(const __bf16* A, int lda, int m0, int k0, int lane) {
  int mr = m0 + (lane & 15), half = lane >> 4;
  const __bf16* base = A + mr * lda + k0 + 16 * half;
  v8bf a0 = *(const v8bf*)(base);
  v8bf a1 = *(const v8bf*)(base + 8);
  return __builtin_shufflevector(a0, a1, 0, 1, 2, 3, 4, 5, 6, 7, 8, 9, 10, 11, 12, 13, 14, 15);
}
// C/D f32 16x16: lane holds col n = n0+(lane&15), rows m = r + 8*(lane>>4)
__device__ __forceinline__ void store_c_frag(float* C, int ldc, int m0, int n0, int lane, v8f c) {
  int n = n0 + (lane & 15);
  int mb = m0 + 8 * (lane >> 4);
#pragma unroll
  for (int r = 0; r < 8; ++r) C[(mb + r) * ldc + n] = c[r];
}
__device__ __forceinline__ void store_c_frag_bf(__bf16* C, int ldc, int m0, int n0, int lane, v8f c) {
  int n = n0 + (lane & 15);
  int mb = m0 + 8 * (lane >> 4);
#pragma unroll
  for (int r = 0; r < 8; ++r) C[(mb + r) * ldc + n] = f2bf(c[r]);
}

// ======================= Kernel 1: trajectory LSTM ==========================
__global__ __launch_bounds__(128) void traj_lstm_kernel(
    const float* __restrict__ obs, const float* __restrict__ h0,
    const float* __restrict__ c0, const float* __restrict__ w_ih,
    const float* __restrict__ w_hh, const float* __restrict__ b_ih,
    const float* __restrict__ b_hh, float* __restrict__ trajhs) {
  __shared__ __align__(32) __bf16 WcP[16 * 512];      // packed: 2 ktiles x 8 ntiles
  __shared__ __align__(32) __bf16 Abuf[4][16][64];    // per wave: [x(3)|h(32)|pad]
  __shared__ float Gbuf[4][16][128];
  const int tid = threadIdx.x, lane = tid & 31, wv = tid >> 5;
  for (int idx = tid; idx < 64 * 128; idx += 128) {
    int k = idx >> 7, n = idx & 127;
    float v = 0.f;
    if (k < 3) v = w_ih[n * 3 + k];
    else if (k < 35) v = w_hh[n * 32 + (k - 3)];
    WcP[b_pack_off(k, n, 8)] = f2bf(v);
  }
  const int base = blockIdx.x * 64 + wv * 16;
  for (int idx = lane; idx < 16 * 64; idx += 32) Abuf[wv][idx >> 6][idx & 63] = f2bf(0.f);
  float hreg[16], creg[16];
#pragma unroll
  for (int a = 0; a < 16; ++a) {
    hreg[a] = h0[(base + a) * HT + lane];
    creg[a] = c0[(base + a) * HT + lane];
    Abuf[wv][a][3 + lane] = f2bf(hreg[a]);
  }
  __syncthreads();
  for (int t = 0; t < OBS; ++t) {
    if (lane < 16) {
#pragma unroll
      for (int cc = 0; cc < 3; ++cc)
        Abuf[wv][lane][cc] = f2bf(obs[((size_t)t * NN + base + lane) * 3 + cc]);
    }
    __syncthreads();
    v16bf a0 = load_a_frag(&Abuf[wv][0][0], 64, 0, 0, lane);
    v16bf a1 = load_a_frag(&Abuf[wv][0][0], 64, 0, 32, lane);
    for (int nt = 0; nt < 8; ++nt) {
      float bv = b_ih[nt * 16 + (lane & 15)] + b_hh[nt * 16 + (lane & 15)];
      v8f acc;
#pragma unroll
      for (int r = 0; r < 8; ++r) acc[r] = bv;
      acc = wmma_bf(a0, load_b_packed(WcP, nt, lane), acc);
      acc = wmma_bf(a1, load_b_packed(WcP, 8 + nt, lane), acc);
      store_c_frag(&Gbuf[wv][0][0], 128, 0, nt * 16, lane, acc);
    }
    __syncthreads();
#pragma unroll
    for (int a = 0; a < 16; ++a) {
      float gi = Gbuf[wv][a][lane], gf = Gbuf[wv][a][32 + lane];
      float gg = Gbuf[wv][a][64 + lane], go = Gbuf[wv][a][96 + lane];
      creg[a] = sigf(gf) * creg[a] + sigf(gi) * tanhf(gg);
      hreg[a] = sigf(go) * tanhf(creg[a]);
      Abuf[wv][a][3 + lane] = f2bf(hreg[a]);
      trajhs[((size_t)t * NN + base + a) * HT + lane] = hreg[a];
    }
    __syncthreads();
  }
}

// ======================= Kernel 2: GAT (per graph,t) ========================
__global__ __launch_bounds__(64) void gat_kernel(
    const float* __restrict__ trajhs, const float* __restrict__ w1,
    const float* __restrict__ a1s, const float* __restrict__ a1d,
    const float* __restrict__ b1, const float* __restrict__ w2,
    const float* __restrict__ a2s, const float* __restrict__ a2d,
    const float* __restrict__ b2, float* __restrict__ gout) {
  __shared__ __align__(32) __bf16 Xb[64][64];
  __shared__ __align__(32) __bf16 WbP[2048];
  __shared__ float HPf[64][64];
  __shared__ __align__(32) __bf16 HPbP[4096];
  __shared__ __align__(32) __bf16 Atb[64][64];
  __shared__ __align__(16) float SDT[8][64];  // [0..3]=s per head, [4..7]=d per head
  const int tid = threadIdx.x, lane = tid & 31, wv = tid >> 5;
  const int t = blockIdx.x >> 10, g = blockIdx.x & 1023;
  const float* xin = trajhs + ((size_t)t * NN + g * 64) * HT;
  // instance-norm over peds, per feature (axis=1 in reference)
  if (tid < 32) {
    int f = tid;
    float m = 0.f;
    for (int p = 0; p < 64; ++p) m += xin[p * 32 + f];
    m *= (1.f / 64.f);
    float v = 0.f;
    for (int p = 0; p < 64; ++p) { float d = xin[p * 32 + f] - m; v += d * d; }
    float rs = rsqrtf(v * (1.f / 64.f) + 1e-5f);
    for (int p = 0; p < 64; ++p) Xb[p][f] = f2bf((xin[p * 32 + f] - m) * rs);
  }
  for (int idx = tid; idx < 32 * 64; idx += 64) {   // W1 packed, NT=4
    int k = idx >> 6, c = idx & 63;
    WbP[b_pack_off(k, c, 4)] = f2bf(w1[((c >> 4) * 32 + k) * 16 + (c & 15)]);
  }
  __syncthreads();
  for (int mt = wv * 2; mt < wv * 2 + 2; ++mt) {     // hp = Xn @ W1 (64x32x64)
    v16bf aF = load_a_frag(&Xb[0][0], 64, mt * 16, 0, lane);
    for (int nt = 0; nt < 4; ++nt) {
      v8f acc;
#pragma unroll
      for (int r = 0; r < 8; ++r) acc[r] = 0.f;
      acc = wmma_bf(aF, load_b_packed(WbP, nt, lane), acc);
      store_c_frag(&HPf[0][0], 64, mt * 16, nt * 16, lane, acc);
    }
  }
  __syncthreads();
  {                                                  // s,d dots + packed bf16 copy
    int p = tid;
#pragma unroll
    for (int h = 0; h < 4; ++h) {
      const float4* hp4 = (const float4*)&HPf[p][h * 16];
      const float4* as4 = (const float4*)&a1s[h * 16];
      const float4* ad4 = (const float4*)&a1d[h * 16];
      float s = 0.f, d = 0.f;
#pragma unroll
      for (int q = 0; q < 4; ++q) {
        float4 hv = hp4[q], as = as4[q], ad = ad4[q];
        s += hv.x * as.x + hv.y * as.y + hv.z * as.z + hv.w * as.w;
        d += hv.x * ad.x + hv.y * ad.y + hv.z * ad.z + hv.w * ad.w;
      }
      SDT[h][p] = s; SDT[4 + h][p] = d;
    }
    for (int c = 0; c < 64; ++c) HPbP[b_pack_off(p, c, 4)] = f2bf(HPf[p][c]);
  }
  __syncthreads();
  for (int hh = 0; hh < 4; ++hh) {                   // per-head softmax + attn@hp
    {
      int p = tid; float s = SDT[hh][p];
      const float4* d4 = (const float4*)&SDT[4 + hh][0];
      float mx = -1e30f;
      for (int q = 0; q < 16; ++q) {
        float4 dv = d4[q];
        float x0 = s + dv.x; x0 = x0 > 0.f ? x0 : 0.2f * x0;
        float x1 = s + dv.y; x1 = x1 > 0.f ? x1 : 0.2f * x1;
        float x2 = s + dv.z; x2 = x2 > 0.f ? x2 : 0.2f * x2;
        float x3 = s + dv.w; x3 = x3 > 0.f ? x3 : 0.2f * x3;
        mx = fmaxf(mx, fmaxf(fmaxf(x0, x1), fmaxf(x2, x3)));
      }
      float sum = 0.f;
      for (int q = 0; q < 16; ++q) {
        float4 dv = d4[q];
        float x0 = s + dv.x; x0 = x0 > 0.f ? x0 : 0.2f * x0;
        float x1 = s + dv.y; x1 = x1 > 0.f ? x1 : 0.2f * x1;
        float x2 = s + dv.z; x2 = x2 > 0.f ? x2 : 0.2f * x2;
        float x3 = s + dv.w; x3 = x3 > 0.f ? x3 : 0.2f * x3;
        sum += __expf(x0 - mx) + __expf(x1 - mx) + __expf(x2 - mx) + __expf(x3 - mx);
      }
      float inv = 1.f / sum;
      for (int q = 0; q < 16; ++q) {
        float4 dv = d4[q];
        float x0 = s + dv.x; x0 = x0 > 0.f ? x0 : 0.2f * x0;
        float x1 = s + dv.y; x1 = x1 > 0.f ? x1 : 0.2f * x1;
        float x2 = s + dv.z; x2 = x2 > 0.f ? x2 : 0.2f * x2;
        float x3 = s + dv.w; x3 = x3 > 0.f ? x3 : 0.2f * x3;
        Atb[p][q * 4 + 0] = f2bf(__expf(x0 - mx) * inv);
        Atb[p][q * 4 + 1] = f2bf(__expf(x1 - mx) * inv);
        Atb[p][q * 4 + 2] = f2bf(__expf(x2 - mx) * inv);
        Atb[p][q * 4 + 3] = f2bf(__expf(x3 - mx) * inv);
      }
    }
    __syncthreads();
    for (int mt = wv * 2; mt < wv * 2 + 2; ++mt) {
      float bv = b1[lane & 15];
      v8f acc;
#pragma unroll
      for (int r = 0; r < 8; ++r) acc[r] = bv;
      acc = wmma_bf(load_a_frag(&Atb[0][0], 64, mt * 16, 0, lane),  load_b_packed(HPbP, hh, lane), acc);
      acc = wmma_bf(load_a_frag(&Atb[0][0], 64, mt * 16, 32, lane), load_b_packed(HPbP, 4 + hh, lane), acc);
      store_c_frag(&HPf[0][0], 64, mt * 16, hh * 16, lane, acc);
    }
    __syncthreads();
  }
  {                                                  // elu + instance-norm -> Xb
    int f = tid; float m = 0.f;
    for (int p = 0; p < 64; ++p) {
      float x = HPf[p][f]; x = x > 0.f ? x : (__expf(x) - 1.f);
      HPf[p][f] = x; m += x;
    }
    m *= (1.f / 64.f);
    float v = 0.f;
    for (int p = 0; p < 64; ++p) { float d = HPf[p][f] - m; v += d * d; }
    float rs = rsqrtf(v * (1.f / 64.f) + 1e-5f);
    for (int p = 0; p < 64; ++p) Xb[p][f] = f2bf((HPf[p][f] - m) * rs);
  }
  for (int idx = tid; idx < 64 * 32; idx += 64)      // W2 packed, NT=2
    WbP[b_pack_off(idx >> 5, idx & 31, 2)] = f2bf(w2[idx]);
  __syncthreads();
  for (int mt = wv * 2; mt < wv * 2 + 2; ++mt) {     // hp2 = X2 @ W2 (64x64x32)
    v16bf a0 = load_a_frag(&Xb[0][0], 64, mt * 16, 0, lane);
    v16bf a1 = load_a_frag(&Xb[0][0], 64, mt * 16, 32, lane);
    for (int nt = 0; nt < 2; ++nt) {
      v8f acc;
#pragma unroll
      for (int r = 0; r < 8; ++r) acc[r] = 0.f;
      acc = wmma_bf(a0, load_b_packed(WbP, nt, lane), acc);
      acc = wmma_bf(a1, load_b_packed(WbP, 2 + nt, lane), acc);
      store_c_frag(&HPf[0][0], 64, mt * 16, nt * 16, lane, acc);
    }
  }
  __syncthreads();
  {
    int p = tid; float s = 0.f, d = 0.f;
    const float4* hp4 = (const float4*)&HPf[p][0];
    const float4* as4 = (const float4*)&a2s[0];
    const float4* ad4 = (const float4*)&a2d[0];
#pragma unroll
    for (int q = 0; q < 8; ++q) {
      float4 hv = hp4[q], as = as4[q], ad = ad4[q];
      s += hv.x * as.x + hv.y * as.y + hv.z * as.z + hv.w * as.w;
      d += hv.x * ad.x + hv.y * ad.y + hv.z * ad.z + hv.w * ad.w;
    }
    SDT[0][p] = s; SDT[1][p] = d;
    for (int o = 0; o < 32; ++o) HPbP[b_pack_off(p, o, 2)] = f2bf(HPf[p][o]);
  }
  __syncthreads();
  {
    int p = tid; float s = SDT[0][p];
    const float4* d4 = (const float4*)&SDT[1][0];
    float mx = -1e30f;
    for (int q = 0; q < 16; ++q) {
      float4 dv = d4[q];
      float x0 = s + dv.x; x0 = x0 > 0.f ? x0 : 0.2f * x0;
      float x1 = s + dv.y; x1 = x1 > 0.f ? x1 : 0.2f * x1;
      float x2 = s + dv.z; x2 = x2 > 0.f ? x2 : 0.2f * x2;
      float x3 = s + dv.w; x3 = x3 > 0.f ? x3 : 0.2f * x3;
      mx = fmaxf(mx, fmaxf(fmaxf(x0, x1), fmaxf(x2, x3)));
    }
    float sum = 0.f;
    for (int q = 0; q < 16; ++q) {
      float4 dv = d4[q];
      float x0 = s + dv.x; x0 = x0 > 0.f ? x0 : 0.2f * x0;
      float x1 = s + dv.y; x1 = x1 > 0.f ? x1 : 0.2f * x1;
      float x2 = s + dv.z; x2 = x2 > 0.f ? x2 : 0.2f * x2;
      float x3 = s + dv.w; x3 = x3 > 0.f ? x3 : 0.2f * x3;
      sum += __expf(x0 - mx) + __expf(x1 - mx) + __expf(x2 - mx) + __expf(x3 - mx);
    }
    float inv = 1.f / sum;
    for (int q = 0; q < 16; ++q) {
      float4 dv = d4[q];
      float x0 = s + dv.x; x0 = x0 > 0.f ? x0 : 0.2f * x0;
      float x1 = s + dv.y; x1 = x1 > 0.f ? x1 : 0.2f * x1;
      float x2 = s + dv.z; x2 = x2 > 0.f ? x2 : 0.2f * x2;
      float x3 = s + dv.w; x3 = x3 > 0.f ? x3 : 0.2f * x3;
      Atb[p][q * 4 + 0] = f2bf(__expf(x0 - mx) * inv);
      Atb[p][q * 4 + 1] = f2bf(__expf(x1 - mx) * inv);
      Atb[p][q * 4 + 2] = f2bf(__expf(x2 - mx) * inv);
      Atb[p][q * 4 + 3] = f2bf(__expf(x3 - mx) * inv);
    }
  }
  __syncthreads();
  float* outp = gout + ((size_t)t * NN + g * 64) * 32;
  for (int mt = wv * 2; mt < wv * 2 + 2; ++mt) {
    v16bf a0 = load_a_frag(&Atb[0][0], 64, mt * 16, 0, lane);
    v16bf a1 = load_a_frag(&Atb[0][0], 64, mt * 16, 32, lane);
    for (int nt = 0; nt < 2; ++nt) {
      float bv = b2[nt * 16 + (lane & 15)];
      v8f acc;
#pragma unroll
      for (int r = 0; r < 8; ++r) acc[r] = bv;
      acc = wmma_bf(a0, load_b_packed(HPbP, nt, lane), acc);
      acc = wmma_bf(a1, load_b_packed(HPbP, 2 + nt, lane), acc);
      store_c_frag(outp, 32, mt * 16, nt * 16, lane, acc);
    }
  }
}

// ======================= Kernel 3: graph LSTM ===============================
__global__ __launch_bounds__(128) void graph_lstm_kernel(
    const float* __restrict__ gin, const float* __restrict__ h0,
    const float* __restrict__ c0, const float* __restrict__ w_ih,
    const float* __restrict__ w_hh, const float* __restrict__ b_ih,
    const float* __restrict__ b_hh, float* __restrict__ ghfin) {
  __shared__ __align__(32) __bf16 WcP[16 * 512];
  __shared__ __align__(32) __bf16 Abuf[4][16][64];
  __shared__ float Gbuf[4][16][128];
  const int tid = threadIdx.x, lane = tid & 31, wv = tid >> 5;
  for (int idx = tid; idx < 64 * 128; idx += 128) {
    int k = idx >> 7, n = idx & 127;
    WcP[b_pack_off(k, n, 8)] = f2bf(k < 32 ? w_ih[n * 32 + k] : w_hh[n * 32 + (k - 32)]);
  }
  const int base = blockIdx.x * 64 + wv * 16;
  float hreg[16], creg[16];
#pragma unroll
  for (int a = 0; a < 16; ++a) {
    hreg[a] = h0[(base + a) * HG + lane];
    creg[a] = c0[(base + a) * HG + lane];
    Abuf[wv][a][32 + lane] = f2bf(hreg[a]);
  }
  __syncthreads();
  for (int t = 0; t < OBS; ++t) {
#pragma unroll
    for (int a = 0; a < 16; ++a)
      Abuf[wv][a][lane] = f2bf(gin[((size_t)t * NN + base + a) * HG + lane]);
    __syncthreads();
    v16bf a0 = load_a_frag(&Abuf[wv][0][0], 64, 0, 0, lane);
    v16bf a1 = load_a_frag(&Abuf[wv][0][0], 64, 0, 32, lane);
    for (int nt = 0; nt < 8; ++nt) {
      float bv = b_ih[nt * 16 + (lane & 15)] + b_hh[nt * 16 + (lane & 15)];
      v8f acc;
#pragma unroll
      for (int r = 0; r < 8; ++r) acc[r] = bv;
      acc = wmma_bf(a0, load_b_packed(WcP, nt, lane), acc);
      acc = wmma_bf(a1, load_b_packed(WcP, 8 + nt, lane), acc);
      store_c_frag(&Gbuf[wv][0][0], 128, 0, nt * 16, lane, acc);
    }
    __syncthreads();
#pragma unroll
    for (int a = 0; a < 16; ++a) {
      float gi = Gbuf[wv][a][lane], gf = Gbuf[wv][a][32 + lane];
      float gg = Gbuf[wv][a][64 + lane], go = Gbuf[wv][a][96 + lane];
      creg[a] = sigf(gf) * creg[a] + sigf(gi) * tanhf(gg);
      hreg[a] = sigf(go) * tanhf(creg[a]);
      Abuf[wv][a][32 + lane] = f2bf(hreg[a]);
    }
    __syncthreads();
  }
#pragma unroll
  for (int a = 0; a < 16; ++a) ghfin[(base + a) * HG + lane] = hreg[a];
}

// ======================= Kernel 4: decoder (dual LSTM) ======================
__global__ __launch_bounds__(32) void dec_kernel(
    const float* __restrict__ obs, const float* __restrict__ z,
    const float* __restrict__ trajhs, const float* __restrict__ ghfin,
    const float* __restrict__ w_ih, const float* __restrict__ w_hh,
    const float* __restrict__ b_ih, const float* __restrict__ b_hh,
    const float* __restrict__ w_out, const float* __restrict__ b_out,
    float* __restrict__ preds) {
  __shared__ __align__(32) __bf16 WcP[2 * 18 * 512]; // 36,864 B : ktiles 0..1 packed
  __shared__ __align__(32) __bf16 WtP[18 * 256];     //  9,216 B : tail rows 64..73,
                                                     //   compact 16-lane tiles (half 0)
  __shared__ __align__(32) __bf16 Ab[16][96];        //  3,072 B : [x(2)|h(72)|pad]
  __shared__ __align__(32) __bf16 Gb[16][288];       //  9,216 B : gates (bf16)
  const int lane = threadIdx.x;
  const int half = lane >> 4, nc = lane & 15;
  const int base = blockIdx.x * 16;
  for (int idx = lane; idx < 64 * 288; idx += 32) {
    int k = idx / 288, n = idx % 288;
    WcP[b_pack_off(k, n, 18)] = f2bf(k < 2 ? w_ih[n * 2 + k] : w_hh[n * 72 + (k - 2)]);
  }
  // tail: tile nt holds, for lane nlow (=n&15), 16 bf16 at k=64+e; e>=10 is zero pad
  for (int idx = lane; idx < 18 * 256; idx += 32) {
    int nt = idx >> 8, rem = idx & 255, nlow = rem >> 4, e = rem & 15;
    float v = (e < 10) ? w_hh[(nt * 16 + nlow) * 72 + 62 + e] : 0.f;
    WtP[idx] = f2bf(v);
  }
  for (int idx = lane; idx < 16 * 96; idx += 32) Ab[idx / 96][idx % 96] = f2bf(0.f);
  float h[36], c[36], hc[36], cc[36];
#pragma unroll
  for (int ii = 0; ii < 36; ++ii) {
    int lin = ii * 32 + lane, a = lin / 72, j = lin - a * 72;
    int n = base + a;
    float hv;
    if (j < 32) hv = trajhs[((size_t)7 * NN + n) * 32 + j];
    else if (j < 64) hv = ghfin[n * 32 + (j - 32)];
    else hv = z[(n >> 6) * 8 + (j - 64)];
    h[ii] = hv;
    hc[ii] = (j < 32) ? 0.f : hv;
    c[ii] = 0.f; cc[ii] = 0.f;
  }
  if (lane < 16) {
    int n = base + lane;
    Ab[lane][0] = f2bf(obs[((size_t)7 * NN + n) * 3 + 0]);
    Ab[lane][1] = f2bf(obs[((size_t)7 * NN + n) * 3 + 1]);
  }
  __syncthreads();
  for (int t = 0; t < FUT; ++t) {
    for (int path = 0; path < 2; ++path) {
#pragma unroll
      for (int ii = 0; ii < 36; ++ii) {
        int lin = ii * 32 + lane, a = lin / 72, j = lin - a * 72;
        Ab[a][2 + j] = f2bf(path == 0 ? h[ii] : hc[ii]);
      }
      __syncthreads();
      v16bf a0 = load_a_frag(&Ab[0][0], 96, 0, 0, lane);
      v16bf a1 = load_a_frag(&Ab[0][0], 96, 0, 32, lane);
      v16bf a2 = load_a_frag(&Ab[0][0], 96, 0, 64, lane);
      for (int nt = 0; nt < 18; ++nt) {
        float bb = b_ih[nt * 16 + nc] + b_hh[nt * 16 + nc];
        v8f acc;
#pragma unroll
        for (int r = 0; r < 8; ++r) acc[r] = bb;
        acc = wmma_bf(a0, load_b_packed(WcP, nt, lane), acc);
        acc = wmma_bf(a1, load_b_packed(WcP, 18 + nt, lane), acc);
        // tail tile: lanes 16..31 read k=80..95 which are all zero
        v16bf bt;
#pragma unroll
        for (int e = 0; e < 16; ++e) bt[e] = f2bf(0.f);
        if (half == 0) bt = *(const v16bf*)(WtP + (nt * 16 + nc) * 16);
        acc = wmma_bf(a2, bt, acc);
        store_c_frag_bf(&Gb[0][0], 288, 0, nt * 16, lane, acc);
      }
      __syncthreads();
#pragma unroll
      for (int ii = 0; ii < 36; ++ii) {
        int lin = ii * 32 + lane, a = lin / 72, j = lin - a * 72;
        float gi = (float)Gb[a][j], gf = (float)Gb[a][72 + j];
        float gg = (float)Gb[a][144 + j], go = (float)Gb[a][216 + j];
        if (path == 0) {
          c[ii] = sigf(gf) * c[ii] + sigf(gi) * tanhf(gg);
          h[ii] = sigf(go) * tanhf(c[ii]);
        } else {
          cc[ii] = sigf(gf) * cc[ii] + sigf(gi) * tanhf(gg);
          hc[ii] = sigf(go) * tanhf(cc[ii]);
        }
      }
      __syncthreads();
    }
#pragma unroll
    for (int ii = 0; ii < 36; ++ii) {
      int lin = ii * 32 + lane, a = lin / 72, j = lin - a * 72;
      Gb[a][j] = f2bf(h[ii] - hc[ii]);
    }
    __syncthreads();
    if (lane < 16) {
      int n = base + lane;
      float o0 = b_out[0], o1 = b_out[1];
      for (int j = 0; j < 72; ++j) {
        float d = (float)Gb[lane][j];
        o0 += d * w_out[j];
        o1 += d * w_out[72 + j];
      }
      preds[((size_t)t * NN + n) * 2 + 0] = o0;
      preds[((size_t)t * NN + n) * 2 + 1] = o1;
      Ab[lane][0] = f2bf(o0);
      Ab[lane][1] = f2bf(o1);
    }
    __syncthreads();
  }
}

extern "C" void kernel_launch(void* const* d_in, const int* in_sizes, int n_in,
                              void* d_out, int out_size, void* d_ws, size_t ws_size,
                              hipStream_t stream) {
  const float* obs  = (const float*)d_in[0];
  const float* z    = (const float*)d_in[1];
  const float* th0  = (const float*)d_in[2];
  const float* tc0  = (const float*)d_in[3];
  const float* gh0  = (const float*)d_in[4];
  const float* gc0  = (const float*)d_in[5];
  const float* wiht = (const float*)d_in[6];
  const float* whht = (const float*)d_in[7];
  const float* biht = (const float*)d_in[8];
  const float* bhht = (const float*)d_in[9];
  const float* g1w  = (const float*)d_in[10];
  const float* g1s  = (const float*)d_in[11];
  const float* g1d  = (const float*)d_in[12];
  const float* g1b  = (const float*)d_in[13];
  const float* g2w  = (const float*)d_in[14];
  const float* g2s  = (const float*)d_in[15];
  const float* g2d  = (const float*)d_in[16];
  const float* g2b  = (const float*)d_in[17];
  const float* wihg = (const float*)d_in[18];
  const float* whhg = (const float*)d_in[19];
  const float* bihg = (const float*)d_in[20];
  const float* bhhg = (const float*)d_in[21];
  const float* wihp = (const float*)d_in[22];
  const float* whhp = (const float*)d_in[23];
  const float* bihp = (const float*)d_in[24];
  const float* bhhp = (const float*)d_in[25];
  const float* wout = (const float*)d_in[26];
  const float* bout = (const float*)d_in[27];
  float* out = (float*)d_out;

  float* ws     = (float*)d_ws;
  float* trajhs = ws;                                 // 8*N*32 floats
  float* gin    = trajhs + (size_t)8 * NN * 32;       // 8*N*32 floats
  float* ghf    = gin    + (size_t)8 * NN * 32;       // N*32 floats

  traj_lstm_kernel<<<NN / 64, 128, 0, stream>>>(obs, th0, tc0, wiht, whht, biht, bhht, trajhs);
  gat_kernel<<<OBS * GG, 64, 0, stream>>>(trajhs, g1w, g1s, g1d, g1b, g2w, g2s, g2d, g2b, gin);
  graph_lstm_kernel<<<NN / 64, 128, 0, stream>>>(gin, gh0, gc0, wihg, whhg, bihg, bhhg, ghf);
  dec_kernel<<<NN / 16, 32, 0, stream>>>(obs, z, trajhs, ghf, wihp, whhp, bihp, bhhp, wout, bout, out);
}
